// DINOLoss_59150289600966
// MI455X (gfx1250) — matrix operand order repaired and checked by reading, not compiled, not run on hardware
//
#include <hip/hip_runtime.h>

// ---------------- problem constants ----------------
#define DDIM    65536
#define NT      128            // teacher rows  (2 crops * 64)
#define NSR     640            // student rows  (10 crops * 64)
#define NSLICE  8
#define KSLICE  (DDIM / NSLICE)   // 8192
#define INV_ST  10.0f          // 1 / STUDENT_TEMP

typedef __attribute__((ext_vector_type(16))) __bf16 v16bf;
typedef __attribute__((ext_vector_type(8)))  __bf16 v8bf;
typedef __attribute__((ext_vector_type(8)))  float  v8f;

__device__ __forceinline__ float teacher_inv_temp(const int* epoch) {
  const int e = *epoch;
  const float tt = (e < 30) ? (0.04f + (0.07f - 0.04f) * (float)e / 30.0f) : 0.07f;
  return 1.0f / tt;
}

// ---- kernel 1: teacher row softmax stats + write normalized P in bf16 ----
__global__ __launch_bounds__(256) void dino_teacher_stats(
    const float* __restrict__ T, const float* __restrict__ center,
    const int* __restrict__ epoch, void* __restrict__ pBFraw) {
  __shared__ float red[256];
  __bf16* pBF = (__bf16*)pBFraw;
  const int i = blockIdx.x;
  const int tid = threadIdx.x;
  const float invtt = teacher_inv_temp(epoch);
  const float* row = T + (size_t)i * DDIM;

  float lmax = -3.4e38f;
  for (int k = tid; k < DDIM; k += 256)
    lmax = fmaxf(lmax, (row[k] - center[k]) * invtt);
  red[tid] = lmax; __syncthreads();
  for (int s = 128; s > 0; s >>= 1) {
    if (tid < s) red[tid] = fmaxf(red[tid], red[tid + s]);
    __syncthreads();
  }
  const float m = red[0];
  __syncthreads();

  float lsum = 0.0f;
  for (int k = tid; k < DDIM; k += 256)
    lsum += __expf((row[k] - center[k]) * invtt - m);
  red[tid] = lsum; __syncthreads();
  for (int s = 128; s > 0; s >>= 1) {
    if (tid < s) red[tid] += red[tid + s];
    __syncthreads();
  }
  const float iz = 1.0f / red[0];

  __bf16* prow = pBF + (size_t)i * DDIM;
  for (int k = tid; k < DDIM; k += 256)
    prow[k] = (__bf16)(__expf((row[k] - center[k]) * invtt - m) * iz);
}

// ---- kernel 2: student log-softmax constant c_v = max + log(sumexp) ----
__global__ __launch_bounds__(256) void dino_student_stats(
    const float* __restrict__ S, float* __restrict__ sC) {
  __shared__ float red[256];
  const int v = blockIdx.x;
  const int tid = threadIdx.x;
  const float* row = S + (size_t)v * DDIM;

  float lmax = -3.4e38f;
  for (int k = tid; k < DDIM; k += 256)
    lmax = fmaxf(lmax, row[k] * INV_ST);
  red[tid] = lmax; __syncthreads();
  for (int s = 128; s > 0; s >>= 1) {
    if (tid < s) red[tid] = fmaxf(red[tid], red[tid + s]);
    __syncthreads();
  }
  const float m = red[0];
  __syncthreads();

  float lsum = 0.0f;
  for (int k = tid; k < DDIM; k += 256)
    lsum += __expf(row[k] * INV_ST - m);
  red[tid] = lsum; __syncthreads();
  for (int s = 128; s > 0; s >>= 1) {
    if (tid < s) red[tid] += red[tid + s];
    __syncthreads();
  }
  if (tid == 0) sC[v] = m + logf(red[0]);
}

// ---- kernel 3: G = P @ S^T via v_wmma_f32_16x16x32_bf16, K-sliced ----
// Grid: x = student col tile (40), z = K slice (8). Block = 256 = 8 waves.
// Wave w computes teacher rows [16w,16w+16) x 16 student cols -> S read once.
__global__ __launch_bounds__(256) void dino_gemm_wmma(
    const void* __restrict__ pBFraw, const float* __restrict__ S,
    float* __restrict__ gPart) {
  const __bf16* pBF = (const __bf16*)pBFraw;
  const int tid  = threadIdx.x;
  const int lane = tid & 31;
  const int wave = tid >> 5;
  const int rowBase = wave << 4;          // 0..112 (covers all 128 teacher rows)
  const int colBase = blockIdx.x << 4;    // 0..624
  const int lo   = lane & 15;
  const int hi   = lane >> 4;             // 0 or 1
  const int half8  = hi << 3;             // A frag K sub-offset (16-bit A layout)
  const int half16 = hi << 4;             // unused; B frag K offset is half8, same as A

  const __bf16* prow = pBF + (size_t)(rowBase + lo) * DDIM;  // A: teacher row
  const float*  srow = S   + (size_t)(colBase + lo) * DDIM;  // B: student row (= col of S^T)

  v8f acc = {};
  const int k0 = blockIdx.z * KSLICE;
  for (int kk = k0; kk < k0 + KSLICE; kk += 32) {
    // A 16x32 bf16: lane<16 -> K {0..7,16..23}, lane>=16 -> K {8..15,24..31}
    v8bf a0 = *(const v8bf*)(prow + kk + half8);
    v8bf a1 = *(const v8bf*)(prow + kk + 16 + half8);
    // B 32x16 bf16: lane<16 -> K {0..7,16..23}, lane>=16 -> K {8..15,24..31}
    float sv[16];
    const float* sp = srow + kk + half8;
    ((float4*)sv)[0] = ((const float4*)sp)[0];
    ((float4*)sv)[1] = ((const float4*)sp)[1];
    ((float4*)sv)[2] = ((const float4*)sp)[4];
    ((float4*)sv)[3] = ((const float4*)sp)[5];

    v16bf a, b;
#pragma unroll
    for (int j = 0; j < 8; ++j) { a[j] = a0[j]; a[j + 8] = a1[j]; }
#pragma unroll
    for (int j = 0; j < 16; ++j) b[j] = (__bf16)sv[j];

    acc = __builtin_amdgcn_wmma_f32_16x16x32_bf16(
        false, a, false, b, (short)0, acc, false, false);
  }

  // C/D f32 layout: VGPR g -> row = rowBase + g + 8*hi, col = colBase + lo
  float* gout = gPart + (size_t)blockIdx.z * (NT * NSR);
  const int orow = rowBase + hi * 8;
  const int ocol = colBase + lo;
#pragma unroll
  for (int g = 0; g < 8; ++g)
    gout[(size_t)(orow + g) * NSR + ocol] = acc[g];
}

// ---- kernel 4: fold K-slices, partial (sum G, trace G) per block ----
__global__ __launch_bounds__(256) void dino_reduce_g(
    const float* __restrict__ gPart, double* __restrict__ blockOut) {
  __shared__ double rs[256];
  __shared__ double rt[256];
  const int tid = threadIdx.x;
  double s = 0.0, t = 0.0;
  const int base = blockIdx.x * 1024;
#pragma unroll
  for (int q = 0; q < 4; ++q) {
    const int idx = base + q * 256 + tid;     // 0 .. NT*NSR-1
    const int i = idx / NSR;
    const int v = idx - i * NSR;
    double g = 0.0;
    for (int z = 0; z < NSLICE; ++z)
      g += (double)gPart[(size_t)z * (NT * NSR) + idx];
    s += g;
    if (v == i) t += g;
  }
  rs[tid] = s; rt[tid] = t; __syncthreads();
  for (int st = 128; st > 0; st >>= 1) {
    if (tid < st) { rs[tid] += rs[tid + st]; rt[tid] += rt[tid + st]; }
    __syncthreads();
  }
  if (tid == 0) {
    blockOut[blockIdx.x * 2 + 0] = rs[0];
    blockOut[blockIdx.x * 2 + 1] = rt[0];
  }
}

// ---- kernel 5: final scalar loss ----
__global__ __launch_bounds__(256) void dino_finalize(
    const double* __restrict__ blockOut, const float* __restrict__ sC,
    float* __restrict__ out) {
  __shared__ double rc[256];
  __shared__ double rh[256];
  const int tid = threadIdx.x;
  double sc = 0.0, sh = 0.0;
  for (int v = tid; v < NSR; v += 256) {
    const double c = (double)sC[v];
    sc += c;
    if (v < NT) sh += c;
  }
  rc[tid] = sc; rh[tid] = sh; __syncthreads();
  for (int st = 128; st > 0; st >>= 1) {
    if (tid < st) { rc[tid] += rc[tid + st]; rh[tid] += rh[tid + st]; }
    __syncthreads();
  }
  if (tid == 0) {
    double sumG = 0.0, trG = 0.0;
    for (int b = 0; b < (NT * NSR) / 1024; ++b) {
      sumG += blockOut[2 * b + 0];
      trG  += blockOut[2 * b + 1];
    }
    // M[i,v] = 10*G[i,v] - c_v  (since sum_k P[i,k] == 1)
    const double sumM = 10.0 * sumG - (double)NT * rc[0];
    const double trM  = 10.0 * trG  - rh[0];
    const double loss = -(sumM - trM) / ((double)NT * (double)(NSR - 1));
    out[0] = (float)loss;
  }
}

// ---------------- launcher ----------------
extern "C" void kernel_launch(void* const* d_in, const int* in_sizes, int n_in,
                              void* d_out, int out_size, void* d_ws, size_t ws_size,
                              hipStream_t stream) {
  (void)in_sizes; (void)n_in; (void)out_size; (void)ws_size;
  const float* S      = (const float*)d_in[0];   // [10*64, 65536]
  const float* T      = (const float*)d_in[1];   // [2*64, 65536]
  const float* center = (const float*)d_in[2];   // [65536]
  const int*   epoch  = (const int*)d_in[3];     // [1]

  char* ws = (char*)d_ws;
  float*  sC       = (float*) (ws + 0);                       //   2,560 B
  void*   pBF      = (void*)  (ws + 4096);                    // 16,777,216 B (bf16 P)
  float*  gPart    = (float*) (ws + 4096 + 16777216);         //  2,621,440 B
  double* blockOut = (double*)(ws + 4096 + 16777216 + 2621440); // 1,280 B

  dino_teacher_stats<<<NT, 256, 0, stream>>>(T, center, epoch, pBF);
  dino_student_stats<<<NSR, 256, 0, stream>>>(S, sC);
  dino_gemm_wmma<<<dim3(NSR / 16, 1, NSLICE), 256, 0, stream>>>(pBF, S, gPart);
  dino_reduce_g<<<(NT * NSR) / 1024, 256, 0, stream>>>(gPart, blockOut);
  dino_finalize<<<1, 256, 0, stream>>>(blockOut, sC, (float*)d_out);
}
